// UserSelfAttention_20693152432624
// MI455X (gfx1250) — matrix operand order repaired and checked by reading, not compiled, not run on hardware
//
#include <hip/hip_runtime.h>
#include <hip/hip_bf16.h>

#define NTOK 16384
#define DIM  128

typedef __attribute__((ext_vector_type(16))) __bf16 v16bf;
typedef __attribute__((ext_vector_type(8)))  float  v8f;
typedef int v4i __attribute__((vector_size(16)));   // matches async builtin param type

// native f32 -> bf16 (RTNE) — emits v_cvt_pk_bf16_f32 pairs on gfx1250
__device__ __forceinline__ __bf16 f2bf(float f) { return (__bf16)f; }

#if __has_builtin(__builtin_amdgcn_global_load_async_to_lds_b128)
#define HAVE_ASYNC_LDS 1
#else
#define HAVE_ASYNC_LDS 0
#endif

#if HAVE_ASYNC_LDS && __has_builtin(__builtin_amdgcn_global_load_async_to_lds_b32)
#define HAVE_ASYNC_UID 1
#else
#define HAVE_ASYNC_UID 0
#endif

__device__ __forceinline__ void wait_async0() {
#if HAVE_ASYNC_LDS
#if __has_builtin(__builtin_amdgcn_s_wait_asynccnt)
    __builtin_amdgcn_s_wait_asynccnt(0);
#else
    asm volatile("s_wait_asynccnt 0x0" ::: "memory");
#endif
#endif
}

// log2(e) folded scales: softmax computed in log2 domain via exp2
#define QSCALE 0.1275191963717297f   /* (1/sqrt(128)) * log2(e) */
#define UBIAS  7.213475204444817f    /* 5.0 * log2(e) */

// ---------------------------------------------------------------------------
// Kernel 1: QKV projection.  y = x @ W^T  (per-wave 16x16 tile, K=128)
//   blockIdx.z: 0=Q (pre-scaled by QSCALE), 1=K, 2=V (stored transposed)
// ---------------------------------------------------------------------------
__global__ void qkv_proj_kernel(const float* __restrict__ x,
                                const float* __restrict__ Wq,
                                const float* __restrict__ Wk,
                                const float* __restrict__ Wv,
                                __bf16* __restrict__ Qb,
                                __bf16* __restrict__ Kb,
                                __bf16* __restrict__ Vt)
{
    const int lane  = threadIdx.x & 31;
    const int tileM = blockIdx.x;
    const int tileN = blockIdx.y;
    const int which = blockIdx.z;
    const float* W = (which == 0) ? Wq : (which == 1) ? Wk : Wv;

    const int n    = lane & 15;
    const int hl   = lane >> 4;
    const int mRow = tileM * 16 + n;
    const int kh   = hl * 8;
    const int nCol = tileN * 16 + n;
    const int kb   = hl * 8;

    v8f acc = {};
    #pragma unroll
    for (int k0 = 0; k0 < DIM; k0 += 32) {
        v16bf a, b;
        const float* xa = x + (size_t)mRow * DIM + k0 + kh;
        #pragma unroll
        for (int i = 0; i < 8; ++i) {
            a[i]     = f2bf(xa[i]);
            a[8 + i] = f2bf(xa[16 + i]);
        }
        const float* wb = W + (size_t)nCol * DIM + k0 + kb;
        #pragma unroll
        for (int i = 0; i < 8; ++i) { b[i] = f2bf(wb[i]); b[8 + i] = f2bf(wb[16 + i]); }
        acc = __builtin_amdgcn_wmma_f32_16x16x32_bf16(false, a, false, b,
                                                      (short)0, acc, false, false);
    }

    const int mb = hl * 8;
    if (which == 2) {
        #pragma unroll
        for (int r = 0; r < 8; ++r)
            Vt[(size_t)(tileN * 16 + n) * NTOK + tileM * 16 + mb + r] = f2bf(acc[r]);
    } else {
        __bf16* outp = (which == 0) ? Qb : Kb;
        const float sc = (which == 0) ? QSCALE : 1.0f;
        #pragma unroll
        for (int r = 0; r < 8; ++r)
            outp[(size_t)(tileM * 16 + mb + r) * DIM + tileN * 16 + n] = f2bf(acc[r] * sc);
    }
}

// ---------------------------------------------------------------------------
// Stage one 64-key tile (K rows, Vt columns, key user-ids) into LDS.
// Async-to-LDS path (ASYNCcnt-tracked) when available; sync fallback otherwise.
// ---------------------------------------------------------------------------
__device__ __forceinline__ void stage_tiles(const __bf16* __restrict__ Kb,
                                            const __bf16* __restrict__ Vt,
                                            const int*    __restrict__ uid,
                                            __bf16* kdst, __bf16* vdst, int* udst,
                                            int kBase, int tid)
{
#if HAVE_ASYNC_LDS
    const __bf16* ksrc = Kb + (size_t)kBase * DIM;
    #pragma unroll
    for (int i = 0; i < 4; ++i) {
        int idx = (tid + i * 256) * 8;              // element offset, 16B chunks
        __builtin_amdgcn_global_load_async_to_lds_b128(
            (v4i*)(ksrc + idx), (v4i*)(kdst + idx), 0, 0);
    }
    #pragma unroll
    for (int i = 0; i < 4; ++i) {
        int j   = tid + i * 256;                    // 0..1023 float4s
        int row = j >> 3;                           // dim 0..127
        int c4  = j & 7;
        __builtin_amdgcn_global_load_async_to_lds_b128(
            (v4i*)(Vt + (size_t)row * NTOK + kBase + c4 * 8),
            (v4i*)(vdst + row * 64 + c4 * 8), 0, 0);
    }
#if HAVE_ASYNC_UID
    if (tid < 64)
        __builtin_amdgcn_global_load_async_to_lds_b32(
            (int*)(uid + kBase + tid), udst + tid, 0, 0);
#else
    if (tid < 64) udst[tid] = uid[kBase + tid];
#endif
#else
    {
        const float4* src = (const float4*)(Kb + (size_t)kBase * DIM);
        float4*       dst = (float4*)kdst;
        #pragma unroll
        for (int i = 0; i < 4; ++i) dst[tid + i * 256] = src[tid + i * 256];
        #pragma unroll
        for (int i = 0; i < 4; ++i) {
            int j   = tid + i * 256;
            int row = j >> 3;
            int c4  = j & 7;
            ((float4*)(vdst + row * 64))[c4] =
                *(const float4*)(Vt + (size_t)row * NTOK + kBase + c4 * 8);
        }
        if (tid < 64) udst[tid] = uid[kBase + tid];
    }
#endif
}

// ---------------------------------------------------------------------------
// Kernel 2: flash attention with same-user bias, double-buffered K/V/uid in LDS.
//   256 threads = 8 waves; wave w owns query rows [blk*128 + w*16, +16).
// ---------------------------------------------------------------------------
__global__ void __launch_bounds__(256)
flash_attn_kernel(const __bf16* __restrict__ Qb,
                  const __bf16* __restrict__ Kb,
                  const __bf16* __restrict__ Vt,
                  const int*    __restrict__ uid,
                  float*        __restrict__ out)
{
    extern __shared__ __align__(16) char smem_raw[];
    __bf16* kbuf = (__bf16*)smem_raw;              // 2 x [64][128]  (32 KB)
    __bf16* vbuf = kbuf + 2 * 64 * DIM;            // 2 x [128][64]  (32 KB)
    __bf16* pbuf = vbuf + 2 * DIM * 64;            // 8 x [16][64]   (16 KB)
    int*    ubuf = (int*)(pbuf + 8 * 16 * 64);     // 2 x [64] ints  (512 B)

    const int tid  = threadIdx.x;
    const int lane = tid & 31;
    const int wid  = tid >> 5;
    const int n    = lane & 15;
    const int hl   = lane >> 4;
    const int mb   = hl * 8;
    const int qRow0 = blockIdx.x * 128 + wid * 16;

    // ---- preload Q fragments (A layout) ----
    v16bf qf[4];
    {
        const __bf16* qbase = Qb + (size_t)(qRow0 + n) * DIM + hl * 8;
        #pragma unroll
        for (int ks = 0; ks < 4; ++ks) {
            union { v16bf v; float4 f[2]; } u;
            u.f[0] = *(const float4*)(qbase + ks * 32);
            u.f[1] = *(const float4*)(qbase + ks * 32 + 16);
            qf[ks] = u.v;
        }
    }
    int uq[8];
    #pragma unroll
    for (int r = 0; r < 8; ++r) uq[r] = uid[qRow0 + mb + r];

    v8f o[8];
    {
        v8f z = {};
        #pragma unroll
        for (int d = 0; d < 8; ++d) o[d] = z;
    }
    float m_i[8], l_i[8];
    #pragma unroll
    for (int r = 0; r < 8; ++r) { m_i[r] = -3.0e38f; l_i[r] = 0.0f; }

    const int nTiles = NTOK / 64;
    stage_tiles(Kb, Vt, uid, kbuf, vbuf, ubuf, 0, tid);   // prefetch tile 0

    for (int it = 0; it < nTiles; ++it) {
        const int cur = it & 1;
        __bf16* k_lds = kbuf + cur * 64 * DIM;
        __bf16* v_lds = vbuf + cur * DIM * 64;
        int*    u_lds = ubuf + cur * 64;

        wait_async0();          // own copies for `cur` have landed
        __syncthreads();        // all threads' copies visible; prior-iter reads done

        if (it + 1 < nTiles)    // overlap next tile's fetch with this tile's math
            stage_tiles(Kb, Vt, uid,
                        kbuf + (cur ^ 1) * 64 * DIM,
                        vbuf + (cur ^ 1) * DIM * 64,
                        ubuf + (cur ^ 1) * 64, (it + 1) * 64, tid);

        // ---- S = Q @ K^T (+ same-user bias), log2 domain ----
        float s[4][8];
        #pragma unroll
        for (int t = 0; t < 4; ++t) {
            v8f acc = {};
            #pragma unroll
            for (int ks = 0; ks < 4; ++ks) {
                union { v16bf v; float4 f[2]; } u;
                const __bf16* kp = k_lds + (t * 16 + n) * DIM + ks * 32 + hl * 8;
                u.f[0] = *(const float4*)(kp);
                u.f[1] = *(const float4*)(kp + 16);
                acc = __builtin_amdgcn_wmma_f32_16x16x32_bf16(false, qf[ks], false, u.v,
                                                              (short)0, acc, false, false);
            }
            const int ku = u_lds[t * 16 + n];
            #pragma unroll
            for (int r = 0; r < 8; ++r)
                s[t][r] = acc[r] + ((uq[r] == ku) ? UBIAS : 0.0f);
        }

        // ---- online softmax (exp2 domain) ----
        float mnew[8], alpha[8];
        #pragma unroll
        for (int r = 0; r < 8; ++r) {
            float v = fmaxf(fmaxf(s[0][r], s[1][r]), fmaxf(s[2][r], s[3][r]));
            #pragma unroll
            for (int mask = 1; mask < 16; mask <<= 1)
                v = fmaxf(v, __shfl_xor(v, mask, 32));
            mnew[r]  = fmaxf(m_i[r], v);
            alpha[r] = exp2f(m_i[r] - mnew[r]);
            m_i[r]   = mnew[r];
        }
        __bf16* p_lds = pbuf + wid * 16 * 64;
        #pragma unroll
        for (int t = 0; t < 4; ++t) {
            #pragma unroll
            for (int r = 0; r < 8; ++r) {
                float p = exp2f(s[t][r] - mnew[r]);
                s[t][r] = p;
                p_lds[(mb + r) * 64 + t * 16 + n] = f2bf(p);
            }
        }
        #pragma unroll
        for (int r = 0; r < 8; ++r) {
            float sum = s[0][r] + s[1][r] + s[2][r] + s[3][r];
            #pragma unroll
            for (int mask = 1; mask < 16; mask <<= 1)
                sum += __shfl_xor(sum, mask, 32);
            l_i[r] = l_i[r] * alpha[r] + sum;
        }
        #pragma unroll
        for (int d = 0; d < 8; ++d)
            #pragma unroll
            for (int r = 0; r < 8; ++r) o[d][r] *= alpha[r];

        __syncthreads();        // P visible (cross-lane through LDS)

        // ---- O += P @ V ----
        v16bf pf[2];
        #pragma unroll
        for (int ks = 0; ks < 2; ++ks) {
            union { v16bf v; float4 f[2]; } u;
            const __bf16* pp = p_lds + n * 64 + ks * 32 + hl * 8;
            u.f[0] = *(const float4*)(pp);
            u.f[1] = *(const float4*)(pp + 16);
            pf[ks] = u.v;
        }
        #pragma unroll
        for (int d = 0; d < 8; ++d) {
            #pragma unroll
            for (int ks = 0; ks < 2; ++ks) {
                union { v16bf v; float4 f[2]; } u;
                const __bf16* vp = v_lds + (d * 16 + n) * 64 + ks * 32 + hl * 8;
                u.f[0] = *(const float4*)(vp);
                u.f[1] = *(const float4*)(vp + 16);
                o[d] = __builtin_amdgcn_wmma_f32_16x16x32_bf16(false, pf[ks], false, u.v,
                                                               (short)0, o[d], false, false);
            }
        }
    }

    // ---- normalize and write out (f32) ----
    #pragma unroll
    for (int r = 0; r < 8; ++r) {
        float inv = 1.0f / l_i[r];
        #pragma unroll
        for (int d = 0; d < 8; ++d)
            out[(size_t)(qRow0 + mb + r) * DIM + d * 16 + n] = o[d][r] * inv;
    }
}

// ---------------------------------------------------------------------------
extern "C" void kernel_launch(void* const* d_in, const int* in_sizes, int n_in,
                              void* d_out, int out_size, void* d_ws, size_t ws_size,
                              hipStream_t stream) {
    const float* x   = (const float*)d_in[0];
    const int*   uid = (const int*)d_in[1];
    const float* Wq  = (const float*)d_in[2];
    const float* Wk  = (const float*)d_in[3];
    const float* Wv  = (const float*)d_in[4];
    float*       out = (float*)d_out;

    __bf16* Qb = (__bf16*)d_ws;                       // [NTOK][DIM] bf16 (pre-scaled)
    __bf16* Kb = Qb + (size_t)NTOK * DIM;             // [NTOK][DIM] bf16
    __bf16* Vt = Kb + (size_t)NTOK * DIM;             // [DIM][NTOK] bf16 (transposed)

    qkv_proj_kernel<<<dim3(NTOK / 16, DIM / 16, 3), 32, 0, stream>>>(
        x, Wq, Wk, Wv, Qb, Kb, Vt);

    const size_t smem = (2 * 64 * DIM + 2 * DIM * 64 + 8 * 16 * 64) * sizeof(__bf16)
                      + 2 * 64 * sizeof(int);         // 80 KB + 512 B
    flash_attn_kernel<<<dim3(NTOK / 128, 1, 1), 256, smem, stream>>>(
        Qb, Kb, Vt, uid, out);
}
